// DSA_RNN_27745488732275
// MI455X (gfx1250) — matrix operand order repaired and checked by reading, not compiled, not run on hardware
//
#include <hip/hip_runtime.h>

// ---------------------------------------------------------------------------
// 2-layer peephole LSTM for MI455X (gfx1250).
//   * bf16 data / f32 accumulate via v_wmma_f32_16x16x32_bf16 (wave32).
//   * Layer-0 input projection hoisted out of the recurrence as one big GEMM.
//   * 64x64 per-wave register blocking: 16 WMMA accumulators, 1.0 b128-load
//     per WMMA -> recurrent weights (36 MB bf16, resident in 192 MB L2) are
//     read exactly once per timestep.
// ---------------------------------------------------------------------------

typedef __attribute__((ext_vector_type(16))) __bf16 v16bf;
typedef __attribute__((ext_vector_type(8)))  __bf16 v8bf;
typedef __attribute__((ext_vector_type(8)))  float  v8f;

namespace {
constexpr int kB = 64;      // batch
constexpr int kT = 128;     // time steps
constexpr int kI = 4096;    // input dim (layer 0)
constexpr int kH = 1024;    // hidden dim
}

// ---------------------------------------------------------------------------
// fp32 -> bf16 conversion
// ---------------------------------------------------------------------------
__global__ void dsa_cvt_bf16(const float* __restrict__ s, __bf16* __restrict__ d, int n) {
    int i = blockIdx.x * blockDim.x + threadIdx.x;
    if (i < n) d[i] = (__bf16)s[i];
}

// ---------------------------------------------------------------------------
// Generic segmented WMMA GEMM:  Z[m, n] = sum_k A_seg[m, k] * W_seg[n, k]
// Up to 3 segments along N, each with its own A (activation) and W (weights),
// both bf16 row-major with K contiguous. Each wave computes a 64x64 tile
// (4x4 wmma accumulators; each A/B fragment reused 4x). M must be a multiple
// of 64, segment widths multiples of 64, K a multiple of 32.
// ---------------------------------------------------------------------------
__global__ __launch_bounds__(256) void dsa_wmma_gemm_seg(
    const __bf16* __restrict__ A0, const __bf16* __restrict__ W0, int n0c,
    const __bf16* __restrict__ A1, const __bf16* __restrict__ W1, int n1c,
    const __bf16* __restrict__ A2, const __bf16* __restrict__ W2, int n2c,
    float* __restrict__ Z, int M, int K)
{
    const int ldz    = n0c + n1c + n2c;
    const int mtiles = M >> 6;
    const int ntiles = ldz >> 6;
    const int wave   = blockIdx.x * (blockDim.x >> 5) + (threadIdx.x >> 5);
    if (wave >= mtiles * ntiles) return;           // uniform per wave

    const int mt   = wave % mtiles;
    const int nt   = wave / mtiles;
    const int ncol = nt << 6;

    const __bf16* A; const __bf16* W; int nloc;
    if (ncol < n0c)            { A = A0; W = W0; nloc = ncol; }
    else if (ncol < n0c + n1c) { A = A1; W = W1; nloc = ncol - n0c; }
    else                       { A = A2; W = W2; nloc = ncol - n0c - n1c; }

    const int lane = threadIdx.x & 31;
    const int r    = lane & 15;
    const int hi   = lane >> 4;

    // A fragment i (16x32 bf16): lane holds row M = mt*64 + i*16 + r;
    // contiguous 16 K elements starting at k + 16*hi (matches B).
    const __bf16* aptr = A + (size_t)((mt << 6) + r) * K + (hi << 4);
    // B fragment j (32x16 bf16): lane holds column N = nloc + j*16 + r,
    // contiguous 16 K elements starting at k + 16*hi.
    const __bf16* wptr = W + (size_t)(nloc + r) * K + (hi << 4);
    const size_t  step16 = (size_t)K << 4;         // advance 16 rows

    v8f acc[4][4];
#pragma unroll
    for (int i = 0; i < 4; ++i)
#pragma unroll
        for (int j = 0; j < 4; ++j)
            acc[i][j] = v8f{};

    for (int k = 0; k < K; k += 32) {
        union { v16bf v; v8bf h[2]; } a[4];
        v16bf bfr[4];
#pragma unroll
        for (int i = 0; i < 4; ++i) {
            a[i].h[0] = *(const v8bf*)(aptr + (size_t)i * step16 + k);
            a[i].h[1] = *(const v8bf*)(aptr + (size_t)i * step16 + k + 8);
        }
#pragma unroll
        for (int j = 0; j < 4; ++j)
            bfr[j] = *(const v16bf*)(wptr + (size_t)j * step16 + k);
#pragma unroll
        for (int i = 0; i < 4; ++i)
#pragma unroll
            for (int j = 0; j < 4; ++j)
                acc[i][j] = __builtin_amdgcn_wmma_f32_16x16x32_bf16(
                    false, a[i].v, false, bfr[j], (short)0, acc[i][j], false, false);
    }

    // C/D layout: VGPR v, lane -> M = mt*64 + i*16 + 8*hi + v, N = ncol + 16*j + r.
#pragma unroll
    for (int i = 0; i < 4; ++i) {
        float* zp = Z + (size_t)((mt << 6) + (i << 4) + (hi << 3)) * ldz + ncol + r;
#pragma unroll
        for (int v = 0; v < 8; ++v) {
#pragma unroll
            for (int j = 0; j < 4; ++j)
                zp[(size_t)v * ldz + 16 * j] = acc[i][j][v];
        }
    }
}

__device__ __forceinline__ float dsa_sigmoid(float x) {
    return 1.0f / (1.0f + __expf(-x));
}

// ---------------------------------------------------------------------------
// Layer-0 pointwise: combine precomputed xz0[t], recurrent z0 (hz|cz), bias.
// z0 row layout per batch: [0,4H) = hz gates i,f,g,o ; [4H,7H) = cz i,f,o.
// ---------------------------------------------------------------------------
__global__ void dsa_pw0(const float* __restrict__ xz0, int t,
                        const float* __restrict__ z0,
                        const float* __restrict__ b0,
                        float* __restrict__ ha32, float* __restrict__ ca32,
                        __bf16* __restrict__ ha16, __bf16* __restrict__ ca16)
{
    int idx = blockIdx.x * blockDim.x + threadIdx.x;
    if (idx >= kB * kH) return;
    int b = idx / kH, h = idx % kH;
    const float* xr = xz0 + ((size_t)(b * kT + t)) * (4 * kH);
    const float* zr = z0 + (size_t)b * (7 * kH);
    float zi = xr[0 * kH + h] + zr[0 * kH + h] + zr[4 * kH + h] + b0[0 * kH + h];
    float zf = xr[1 * kH + h] + zr[1 * kH + h] + zr[5 * kH + h] + b0[1 * kH + h];
    float zg = xr[2 * kH + h] + zr[2 * kH + h]                  + b0[2 * kH + h];
    float zo = xr[3 * kH + h] + zr[3 * kH + h] + zr[6 * kH + h] + b0[3 * kH + h];
    float ig = dsa_sigmoid(zi);
    float fg = dsa_sigmoid(zf);
    float gg = tanhf(zg);
    float og = dsa_sigmoid(zo);           // o-gate peeps at OLD c (already in zr)
    float c  = ca32[idx];
    float nc = fg * c + ig * gg;
    float nh = og * tanhf(nc);
    ca32[idx] = nc; ha32[idx] = nh;
    ca16[idx] = (__bf16)nc; ha16[idx] = (__bf16)nh;
}

// ---------------------------------------------------------------------------
// Layer-1 pointwise. z1 row layout per batch:
// [0,4H) xz (Wx1·ha) ; [4H,8H) hz (Uh1·hb) ; [8H,11H) cz (Vc1·cb, i,f,o).
// Writes output[b, t, h].
// ---------------------------------------------------------------------------
__global__ void dsa_pw1(int t,
                        const float* __restrict__ z1,
                        const float* __restrict__ b1,
                        float* __restrict__ hb32, float* __restrict__ cb32,
                        __bf16* __restrict__ hb16, __bf16* __restrict__ cb16,
                        float* __restrict__ out)
{
    int idx = blockIdx.x * blockDim.x + threadIdx.x;
    if (idx >= kB * kH) return;
    int b = idx / kH, h = idx % kH;
    const float* zr = z1 + (size_t)b * (11 * kH);
    float zi = zr[0 * kH + h] + zr[4 * kH + h] + zr[ 8 * kH + h] + b1[0 * kH + h];
    float zf = zr[1 * kH + h] + zr[5 * kH + h] + zr[ 9 * kH + h] + b1[1 * kH + h];
    float zg = zr[2 * kH + h] + zr[6 * kH + h]                   + b1[2 * kH + h];
    float zo = zr[3 * kH + h] + zr[7 * kH + h] + zr[10 * kH + h] + b1[3 * kH + h];
    float ig = dsa_sigmoid(zi);
    float fg = dsa_sigmoid(zf);
    float gg = tanhf(zg);
    float og = dsa_sigmoid(zo);
    float c  = cb32[idx];
    float nc = fg * c + ig * gg;
    float nh = og * tanhf(nc);
    cb32[idx] = nc; hb32[idx] = nh;
    cb16[idx] = (__bf16)nc; hb16[idx] = (__bf16)nh;
    out[(size_t)(b * kT + t) * kH + h] = nh;
}

// ---------------------------------------------------------------------------
// State init from h0/c0 (shape (L,B,H)) and final h_n/c_n copy into d_out tail.
// ---------------------------------------------------------------------------
__global__ void dsa_init_state(const float* __restrict__ h0, const float* __restrict__ c0,
                               float* ha32, float* ca32, float* hb32, float* cb32,
                               __bf16* ha16, __bf16* ca16, __bf16* hb16, __bf16* cb16)
{
    int idx = blockIdx.x * blockDim.x + threadIdx.x;
    const int n = kB * kH;
    if (idx >= n) return;
    float ha = h0[idx], hb = h0[n + idx];
    float ca = c0[idx], cb = c0[n + idx];
    ha32[idx] = ha; hb32[idx] = hb; ca32[idx] = ca; cb32[idx] = cb;
    ha16[idx] = (__bf16)ha; hb16[idx] = (__bf16)hb;
    ca16[idx] = (__bf16)ca; cb16[idx] = (__bf16)cb;
}

__global__ void dsa_finalize(const float* __restrict__ ha32, const float* __restrict__ hb32,
                             const float* __restrict__ ca32, const float* __restrict__ cb32,
                             float* __restrict__ out)
{
    int idx = blockIdx.x * blockDim.x + threadIdx.x;
    const int n = kB * kH;
    if (idx >= n) return;
    float* tail = out + (size_t)kB * kT * kH;
    tail[idx]         = ha32[idx];   // h_n layer 0
    tail[n + idx]     = hb32[idx];   // h_n layer 1
    tail[2 * n + idx] = ca32[idx];   // c_n layer 0
    tail[3 * n + idx] = cb32[idx];   // c_n layer 1
}

// ---------------------------------------------------------------------------
// Host-side launch
// ---------------------------------------------------------------------------
extern "C" void kernel_launch(void* const* d_in, const int* in_sizes, int n_in,
                              void* d_out, int out_size, void* d_ws, size_t ws_size,
                              hipStream_t stream) {
    (void)in_sizes; (void)n_in; (void)out_size; (void)ws_size;
    const float* x   = (const float*)d_in[0];   // (B,T,I)
    const float* h0  = (const float*)d_in[1];   // (L,B,H)
    const float* c0  = (const float*)d_in[2];
    const float* Wx0 = (const float*)d_in[3];   // (4,H,I)
    const float* Uh0 = (const float*)d_in[4];   // (4,H,H)
    const float* Vc0 = (const float*)d_in[5];   // (3,H,H)
    const float* b0  = (const float*)d_in[6];   // (4,H)
    const float* Wx1 = (const float*)d_in[7];   // (4,H,H)
    const float* Uh1 = (const float*)d_in[8];
    const float* Vc1 = (const float*)d_in[9];
    const float* b1  = (const float*)d_in[10];
    float* out = (float*)d_out;

    // Workspace carve-out (256B aligned regions).
    char* ws = (char*)d_ws;
    size_t off = 0;
    auto carve = [&](size_t bytes) -> void* {
        void* p = ws + off;
        off += (bytes + 255) & ~(size_t)255;
        return p;
    };
    const size_t nX   = (size_t)kB * kT * kI;       // 33.5M
    const size_t nWx0 = (size_t)4 * kH * kI;        // 16.8M
    const size_t nHH4 = (size_t)4 * kH * kH;        // 4.2M
    const size_t nHH3 = (size_t)3 * kH * kH;        // 3.1M
    const size_t nBH  = (size_t)kB * kH;            // 64K

    float*  xz0  = (float*) carve((size_t)kB * kT * 4 * kH * sizeof(float)); // (B*T, 4H)
    __bf16* xb   = (__bf16*)carve(nX   * sizeof(__bf16));
    __bf16* wx0b = (__bf16*)carve(nWx0 * sizeof(__bf16));
    __bf16* uh0b = (__bf16*)carve(nHH4 * sizeof(__bf16));
    __bf16* vc0b = (__bf16*)carve(nHH3 * sizeof(__bf16));
    __bf16* wx1b = (__bf16*)carve(nHH4 * sizeof(__bf16));
    __bf16* uh1b = (__bf16*)carve(nHH4 * sizeof(__bf16));
    __bf16* vc1b = (__bf16*)carve(nHH3 * sizeof(__bf16));
    float*  ha32 = (float*) carve(nBH * sizeof(float));
    float*  ca32 = (float*) carve(nBH * sizeof(float));
    float*  hb32 = (float*) carve(nBH * sizeof(float));
    float*  cb32 = (float*) carve(nBH * sizeof(float));
    __bf16* ha16 = (__bf16*)carve(nBH * sizeof(__bf16));
    __bf16* ca16 = (__bf16*)carve(nBH * sizeof(__bf16));
    __bf16* hb16 = (__bf16*)carve(nBH * sizeof(__bf16));
    __bf16* cb16 = (__bf16*)carve(nBH * sizeof(__bf16));
    float*  z0   = (float*) carve((size_t)kB *  7 * kH * sizeof(float));
    float*  z1   = (float*) carve((size_t)kB * 11 * kH * sizeof(float));

    const int TPB = 256;
    auto cvt = [&](const float* s, __bf16* d, size_t n) {
        dsa_cvt_bf16<<<dim3((unsigned)((n + TPB - 1) / TPB)), dim3(TPB), 0, stream>>>(s, d, (int)n);
    };
    cvt(x,   xb,   nX);
    cvt(Wx0, wx0b, nWx0);
    cvt(Uh0, uh0b, nHH4);
    cvt(Vc0, vc0b, nHH3);
    cvt(Wx1, wx1b, nHH4);
    cvt(Uh1, uh1b, nHH4);
    cvt(Vc1, vc1b, nHH3);

    dsa_init_state<<<dim3((kB * kH + TPB - 1) / TPB), dim3(TPB), 0, stream>>>(
        h0, c0, ha32, ca32, hb32, cb32, ha16, ca16, hb16, cb16);

    // Big time-independent GEMM: xz0 = x · Wx0^T  (M=8192, N=4096, K=4096)
    {
        int M = kB * kT, N = 4 * kH, K = kI;
        int waves = (M / 64) * (N / 64);           // 8192
        dsa_wmma_gemm_seg<<<dim3(waves / 8), dim3(TPB), 0, stream>>>(
            xb, wx0b, N, xb, wx0b, 0, xb, wx0b, 0, xz0, M, K);
    }

    // Sequential recurrence.
    const int pwBlocks = (kB * kH + TPB - 1) / TPB;
    for (int t = 0; t < kT; ++t) {
        // layer 0 recurrent gates: z0 = [h·Uh0^T (4H) | c·Vc0^T (3H)]
        {
            int waves = (kB / 64) * ((7 * kH) / 64);   // 112
            dsa_wmma_gemm_seg<<<dim3(waves / 8), dim3(TPB), 0, stream>>>(
                ha16, uh0b, 4 * kH, ca16, vc0b, 3 * kH, ca16, vc0b, 0,
                z0, kB, kH);
        }
        dsa_pw0<<<dim3(pwBlocks), dim3(TPB), 0, stream>>>(
            xz0, t, z0, b0, ha32, ca32, ha16, ca16);

        // layer 1 gates: z1 = [ha·Wx1^T (4H) | hb·Uh1^T (4H) | cb·Vc1^T (3H)]
        {
            int waves = (kB / 64) * ((11 * kH) / 64);  // 176
            dsa_wmma_gemm_seg<<<dim3(waves / 8), dim3(TPB), 0, stream>>>(
                ha16, wx1b, 4 * kH, hb16, uh1b, 4 * kH, cb16, vc1b, 3 * kH,
                z1, kB, kH);
        }
        dsa_pw1<<<dim3(pwBlocks), dim3(TPB), 0, stream>>>(
            t, z1, b1, hb32, cb32, hb16, cb16, out);
    }

    dsa_finalize<<<dim3(pwBlocks), dim3(TPB), 0, stream>>>(ha32, hb32, ca32, cb32, out);
}
